// self_Attention_Block_50766513439214
// MI455X (gfx1250) — matrix operand order repaired and checked by reading, not compiled, not run on hardware
//
#include <hip/hip_runtime.h>

typedef __attribute__((ext_vector_type(16))) _Float16 v16h;
typedef __attribute__((ext_vector_type(8)))  float    v8f;

#define BATCH 2048
#define JNT   17
#define CDIM  512
#define HEADS 8
#define HD    64
#define ON    64
#define HID   2048

// ---- workspace layout (bytes) ----
static constexpr size_t OFF_QWH  = 0;
static constexpr size_t OFF_KVWH = OFF_QWH  + (size_t)512*512*2;
static constexpr size_t OFF_WTH  = OFF_KVWH + (size_t)1024*512*2;
static constexpr size_t OFF_PROJ = OFF_WTH  + (size_t)512*512*2;
static constexpr size_t OFF_FC1  = OFF_PROJ + (size_t)512*512*2;
static constexpr size_t OFF_FC2  = OFF_FC1  + (size_t)2048*512*2;
static constexpr size_t OFF_STAT = OFF_FC2  + (size_t)512*2048*2;
static constexpr size_t OFF_QH   = OFF_STAT + (size_t)1024*4;
static constexpr size_t OFF_KH   = OFF_QH   + (size_t)BATCH*JNT*CDIM*2;
static constexpr size_t OFF_VT   = OFF_KH   + (size_t)BATCH*ON*CDIM*2;
static constexpr size_t OFF_A2   = OFF_VT   + (size_t)BATCH*CDIM*ON*2;
static constexpr size_t OFF_XR   = OFF_A2   + (size_t)BATCH*JNT*CDIM*4;

// adjacency bitmasks (I + skeleton edges, symmetric)
__constant__ unsigned ADJM[17] = {
  0x7u, 0xFu, 0x17u, 0x2Au, 0x54u, 0x8E8u, 0x1170u, 0x2A0u, 0x540u,
  0x280u, 0x500u, 0x3820u, 0x5840u, 0xA800u, 0x15000u, 0xA000u, 0x14000u };

// ---------------- WMMA fragment helpers (wave32, ISA 7.12.2 layouts) ----------------
__device__ __forceinline__ v16h load_a(const _Float16* buf, int lda, int m0, int k0, int lane) {
  int m  = lane & 15;
  int hi = (lane >> 4) & 1;
  const _Float16* p = buf + (size_t)(m0 + m) * lda + k0 + hi * 8;
  v16h a;
#pragma unroll
  for (int j = 0; j < 4; ++j) { a[2*j] = p[2*j]; a[2*j+1] = p[2*j+1]; }
  const _Float16* p2 = p + 16;
#pragma unroll
  for (int j = 0; j < 4; ++j) { a[8+2*j] = p2[2*j]; a[8+2*j+1] = p2[2*j+1]; }
  return a;
}

__device__ __forceinline__ v16h load_b(const _Float16* w, int ldw, int n0, int k0, int lane) {
  int n  = lane & 15;
  int hi = (lane >> 4) & 1;
  const _Float16* p = w + (size_t)(n0 + n) * ldw + k0 + hi * 8;
  v16h b;
#pragma unroll
  for (int j = 0; j < 4; ++j) { b[2*j] = p[2*j]; b[2*j+1] = p[2*j+1]; }
  const _Float16* p2 = p + 16;
#pragma unroll
  for (int j = 0; j < 4; ++j) { b[8+2*j] = p2[2*j]; b[8+2*j+1] = p2[2*j+1]; }
  return b;
}

// C/D 16x16 f32: vgpr r, lane L -> m = r + 8*(L>=16), n = L&15
__device__ __forceinline__ v8f wmma_tile(const _Float16* A, int lda, int m0, int ka0,
                                         const _Float16* Bw, int ldb, int n0, int kb0,
                                         int klen, int lane, v8f c) {
  for (int kk = 0; kk < klen; kk += 32) {
    v16h a = load_a(A, lda, m0, ka0 + kk, lane);
    v16h b = load_b(Bw, ldb, n0, kb0 + kk, lane);
    c = __builtin_amdgcn_wmma_f32_16x16x32_f16(false, a, false, b, (short)0, c, false, false);
  }
  return c;
}

__device__ __forceinline__ float wave_sum(float v) {
#pragma unroll
  for (int off = 16; off > 0; off >>= 1) v += __shfl_xor(v, off, 32);
  return v;
}

// LayerNorm one 512-wide row with one wave, emit f16 into LDS
__device__ __forceinline__ void ln_row(const float* row, _Float16* out,
                                       const float* w, const float* bias, int lane) {
  float vals[16]; float s = 0.f;
#pragma unroll
  for (int t = 0; t < 16; ++t) { vals[t] = row[lane + 32*t]; s += vals[t]; }
  s = wave_sum(s);
  float mu = s * (1.0f/512.0f);
  float q = 0.f;
#pragma unroll
  for (int t = 0; t < 16; ++t) { float d = vals[t]-mu; q += d*d; }
  q = wave_sum(q);
  float rs = rsqrtf(q * (1.0f/512.0f) + 1e-5f);
#pragma unroll
  for (int t = 0; t < 16; ++t) {
    int c = lane + 32*t;
    out[c] = (_Float16)((vals[t]-mu)*rs*w[c] + bias[c]);
  }
}

// ---------------- Kernel 1: weight f32->f16 (+ W transpose) + zero BN stats ----------------
__global__ void __launch_bounds__(256) prep_kernel(
    const float* qw, const float* kvw, const float* W, const float* projw,
    const float* fc1w, const float* fc2w,
    _Float16* qwh, _Float16* kvwh, _Float16* Wth, _Float16* projh,
    _Float16* fc1h, _Float16* fc2h, float* stats) {
  int gid = blockIdx.x * blockDim.x + threadIdx.x;
  const int total = 262144 + 524288 + 262144 + 262144 + 1048576 + 1048576;
  for (int t = gid; t < total; t += gridDim.x * blockDim.x) {
    int idx = t;
    if (idx < 262144) { qwh[idx] = (_Float16)qw[idx]; continue; }
    idx -= 262144;
    if (idx < 524288) { kvwh[idx] = (_Float16)kvw[idx]; continue; }
    idx -= 524288;
    if (idx < 262144) { int n = idx >> 9, k = idx & 511;
                        Wth[idx] = (_Float16)W[k*512 + n]; continue; }
    idx -= 262144;
    if (idx < 262144) { projh[idx] = (_Float16)projw[idx]; continue; }
    idx -= 262144;
    if (idx < 1048576) { fc1h[idx] = (_Float16)fc1w[idx]; continue; }
    idx -= 1048576;
    fc2h[idx] = (_Float16)fc2w[idx];
  }
  if (gid < 1024) stats[gid] = 0.0f;
}

// ---------------- Kernel 2: LN(noise) + Q projection -> qh f16 [B,17,512] ----------------
__global__ void __launch_bounds__(256) qproj_kernel(
    const float* noise, const float* ln1w, const float* ln1b,
    const _Float16* qwh, _Float16* qh) {
  __shared__ _Float16 xl[32*512];
  int b = blockIdx.x, tid = threadIdx.x, wave = tid >> 5, lane = tid & 31;
  for (int i = tid; i < 15*512; i += 256) xl[17*512 + i] = (_Float16)0.f;
  for (int r = wave; r < JNT; r += 8)
    ln_row(noise + ((size_t)b*JNT + r)*CDIM, xl + r*512, ln1w, ln1b, lane);
  __syncthreads();
  int hi8 = (lane >> 4) << 3;
  for (int t = wave; t < 64; t += 8) {
    int mt = t >> 5, nt = t & 31;
    v8f c = {};
    c = wmma_tile(xl, 512, mt*16, 0, qwh, 512, nt*16, 0, 512, lane, c);
    int n = nt*16 + (lane & 15);
    if (mt == 0) {                               // rows 0..15 all valid
      _Float16* dst = qh + (size_t)b*JNT*CDIM + n;
#pragma unroll
      for (int r = 0; r < 8; ++r) dst[(size_t)(r + hi8)*CDIM] = (_Float16)c[r];
    } else if (hi8 == 0) {                       // only row 16 valid
      qh[((size_t)b*JNT + 16)*CDIM + n] = (_Float16)c[0];
    }
  }
}

// ------- Kernel 3: LN(obj) + KV projection -> kh [B,64,512], vt [B,512,64] (transposed) -------
__global__ void __launch_bounds__(256) kvproj_kernel(
    const float* obj, const float* ln1w, const float* ln1b,
    const _Float16* kvwh, _Float16* kh, _Float16* vt) {
  __shared__ _Float16 el[64*512];
  int b = blockIdx.x, tid = threadIdx.x, wave = tid >> 5, lane = tid & 31;
  for (int r = wave; r < ON; r += 8)
    ln_row(obj + ((size_t)b*ON + r)*CDIM, el + r*512, ln1w, ln1b, lane);
  __syncthreads();
  int hi8 = (lane >> 4) << 3;
  for (int t = wave; t < 256; t += 8) {          // 64x1024 output
    int mt = t >> 6, nt = t & 63;
    v8f c = {};
    c = wmma_tile(el, 512, mt*16, 0, kvwh, 512, nt*16, 0, 512, lane, c);
    int nn = nt*16 + (lane & 15);
    if (nt < 32) {                               // k half, rows all valid
      _Float16* dst = kh + (size_t)b*ON*CDIM + (size_t)(mt*16 + hi8)*CDIM + nn;
#pragma unroll
      for (int r = 0; r < 8; ++r) dst[(size_t)r*CDIM] = (_Float16)c[r];
    } else {                                     // v half, store transposed
      _Float16* dst = vt + ((size_t)b*CDIM + (nn - 512))*ON + mt*16 + hi8;
#pragma unroll
      for (int r = 0; r < 8; ++r) dst[r] = (_Float16)c[r];
    }
  }
}

// ------- Kernel 4: per-head QK^T*scale -> [17,512] -> @W^T' -> attn2 f32 + BN stat atomics -------
__global__ void __launch_bounds__(256) attnW_kernel(
    const _Float16* qh, const _Float16* kh, const _Float16* Wth,
    float* attn2, float* stats) {
  __shared__ _Float16 sl[32*512];                 // scores, f16
  __shared__ __align__(16) char ubuf[32*512*4];   // first q f16 (32KB), then attn2 f32 (64KB)
  _Float16* ql = (_Float16*)ubuf;
  float*    ab = (float*)ubuf;
  int b = blockIdx.x, tid = threadIdx.x, wave = tid >> 5, lane = tid & 31;
  for (int i = tid; i < 17*512; i += 256) ql[i] = qh[(size_t)b*JNT*CDIM + i];
  for (int i = tid; i < 15*512; i += 256) ql[17*512 + i] = (_Float16)0.f;
  __syncthreads();
  const _Float16* khb = kh + (size_t)b*ON*CDIM;
  const float SC = 0.125f;                        // 64^-0.5
  int hi8 = (lane >> 4) << 3;
  for (int t = wave; t < 64; t += 8) {            // heads(8) x mt(2) x nt4(4)
    int h = t >> 3, mt = (t >> 2) & 1, nt4 = t & 3;
    v8f c = {};
    c = wmma_tile(ql, 512, mt*16, h*64, khb, 512, nt4*16, h*64, 64, lane, c);
    int j = nt4*16 + (lane & 15);
    _Float16* dst = sl + (size_t)(mt*16 + hi8)*512 + h*64 + j;
#pragma unroll
    for (int r = 0; r < 8; ++r) dst[r*512] = (_Float16)(c[r]*SC);
  }
  __syncthreads();
  for (int t = wave; t < 64; t += 8) {            // [32,512] @ Wt
    int mt = t >> 5, nt = t & 31;
    v8f c = {};
    c = wmma_tile(sl, 512, mt*16, 0, Wth, 512, nt*16, 0, 512, lane, c);
    int n = nt*16 + (lane & 15);
    float* lb = ab + (size_t)(mt*16 + hi8)*512 + n;
#pragma unroll
    for (int r = 0; r < 8; ++r) lb[r*512] = c[r];
    if (mt == 0) {
      float* dst = attn2 + (size_t)b*JNT*CDIM + n;
#pragma unroll
      for (int r = 0; r < 8; ++r) dst[(size_t)(r + hi8)*CDIM] = c[r];
    } else if (hi8 == 0) {
      attn2[((size_t)b*JNT + 16)*CDIM + n] = c[0];
    }
  }
  __syncthreads();
  for (int n = tid; n < 512; n += 256) {
    float s1 = 0.f, s2 = 0.f;
    for (int i = 0; i < JNT; ++i) { float v = ab[i*512 + n]; s1 += v; s2 += v*v; }
    atomicAdd(&stats[n], s1);
    atomicAdd(&stats[512 + n], s2);
  }
}

// ------- Kernel 5: BN apply + adjacency + softmax + @V + proj + residual -> xres f32 -------
__global__ void __launch_bounds__(256) bnattn_kernel(
    const float* attn2, const float* stats, const float* bng, const float* bnb,
    const _Float16* vt, const _Float16* projh, const float* projb,
    const float* noise, float* xres) {
  __shared__ float    nb[17*512];
  __shared__ _Float16 at[32*512];
  __shared__ _Float16 xh[32*512];
  __shared__ float    cs[512], cb[512];
  int b = blockIdx.x, tid = threadIdx.x, wave = tid >> 5, lane = tid & 31;
  const float invN = 1.0f / (float)(BATCH * JNT);
  for (int n = tid; n < 512; n += 256) {
    float mu  = stats[n] * invN;
    float var = stats[512 + n] * invN - mu*mu;
    float sc  = bng[n] * rsqrtf(var + 1e-5f);
    cs[n] = sc; cb[n] = bnb[n] - mu*sc;
  }
  __syncthreads();
  for (int i = tid; i < 17*512; i += 256) {
    int n = i & 511;
    nb[i] = attn2[(size_t)b*JNT*CDIM + i]*cs[n] + cb[n];
  }
  __syncthreads();
  float acc[34]; int cnt = 0;                      // adjacency bmm
  for (int i = tid; i < 17*512; i += 256) {
    int row = i >> 9, n = i & 511;
    unsigned mask = ADJM[row];
    float s = 0.f;
    for (int j = 0; j < JNT; ++j) if (mask & (1u << j)) s += nb[j*512 + n];
    acc[cnt++] = s;
  }
  __syncthreads();
  cnt = 0;
  for (int i = tid; i < 17*512; i += 256) nb[i] = acc[cnt++];
  __syncthreads();
  for (int i = tid; i < 15*512; i += 256) at[17*512 + i] = (_Float16)0.f;
  for (int g = tid; g < JNT*HEADS; g += 256) {     // softmax over 64 per (row, head)
    int row = g >> 3, h = g & 7;
    float* p = nb + row*512 + h*64;
    float mx = -1e30f;
    for (int j = 0; j < 64; ++j) mx = fmaxf(mx, p[j]);
    float sum = 0.f;
    for (int j = 0; j < 64; ++j) { float e = __expf(p[j]-mx); sum += e; p[j] = e; }
    float inv = 1.0f / sum;
    for (int j = 0; j < 64; ++j) at[row*512 + h*64 + j] = (_Float16)(p[j]*inv);
  }
  __syncthreads();
  const _Float16* vtb = vt + (size_t)b*CDIM*ON;
  int hi8 = (lane >> 4) << 3;
  for (int t = wave; t < 64; t += 8) {             // attn @ v per head
    int h = t >> 3, mt = (t >> 2) & 1, nt4 = t & 3;
    v8f c = {};
    c = wmma_tile(at, 512, mt*16, h*64, vtb, 64, h*64 + nt4*16, 0, 64, lane, c);
    int d = nt4*16 + (lane & 15);
    _Float16* dst = xh + (size_t)(mt*16 + hi8)*512 + h*64 + d;
#pragma unroll
    for (int r = 0; r < 8; ++r) dst[r*512] = (_Float16)c[r];
  }
  __syncthreads();
  for (int t = wave; t < 64; t += 8) {             // proj + bias + residual
    int mt = t >> 5, nt = t & 31;
    v8f c = {};
    c = wmma_tile(xh, 512, mt*16, 0, projh, 512, nt*16, 0, 512, lane, c);
    int n = nt*16 + (lane & 15);
    if (mt == 0) {
      size_t o0 = (size_t)b*JNT*CDIM + n;
#pragma unroll
      for (int r = 0; r < 8; ++r) {
        size_t o = o0 + (size_t)(r + hi8)*CDIM;
        xres[o] = c[r] + projb[n] + noise[o];
      }
    } else if (hi8 == 0) {
      size_t o = ((size_t)b*JNT + 16)*CDIM + n;
      xres[o] = c[0] + projb[n] + noise[o];
    }
  }
}

// ------- Kernel 6: fused LN2 + FC1 + GELU + FC2 + residual -> out (hidden stays in LDS) -------
__global__ void __launch_bounds__(256) mlp_kernel(
    const float* xres, const float* ln2w, const float* ln2b,
    const _Float16* fc1h, const float* fc1b,
    const _Float16* fc2h, const float* fc2b, float* out) {
  __shared__ _Float16 xl[32*512];
  __shared__ _Float16 hb[32*2048];
  int b = blockIdx.x, tid = threadIdx.x, wave = tid >> 5, lane = tid & 31;
  for (int i = tid; i < 15*512; i += 256) xl[17*512 + i] = (_Float16)0.f;
  for (int r = wave; r < JNT; r += 8)
    ln_row(xres + ((size_t)b*JNT + r)*CDIM, xl + r*512, ln2w, ln2b, lane);
  __syncthreads();
  int hi8 = (lane >> 4) << 3;
  for (int t = wave; t < 256; t += 8) {            // fc1: [32,512] @ [2048,512]^T
    int mt = t >> 7, nt = t & 127;
    v8f c = {};
    c = wmma_tile(xl, 512, mt*16, 0, fc1h, 512, nt*16, 0, 512, lane, c);
    int n = nt*16 + (lane & 15);
    _Float16* dst = hb + (size_t)(mt*16 + hi8)*2048 + n;
#pragma unroll
    for (int r = 0; r < 8; ++r) {
      float v = c[r] + fc1b[n];
      v = 0.5f * v * (1.0f + erff(v * 0.70710678f));   // exact gelu
      dst[r*2048] = (_Float16)v;
    }
  }
  __syncthreads();
  for (int t = wave; t < 64; t += 8) {             // fc2: [32,2048] @ [512,2048]^T
    int mt = t >> 5, nt = t & 31;
    v8f c = {};
    c = wmma_tile(hb, 2048, mt*16, 0, fc2h, 2048, nt*16, 0, 2048, lane, c);
    int n = nt*16 + (lane & 15);
    if (mt == 0) {
      size_t o0 = (size_t)b*JNT*CDIM + n;
#pragma unroll
      for (int r = 0; r < 8; ++r) {
        size_t o = o0 + (size_t)(r + hi8)*CDIM;
        out[o] = c[r] + fc2b[n] + xres[o];
      }
    } else if (hi8 == 0) {
      size_t o = ((size_t)b*JNT + 16)*CDIM + n;
      out[o] = c[0] + fc2b[n] + xres[o];
    }
  }
}

extern "C" void kernel_launch(void* const* d_in, const int* in_sizes, int n_in,
                              void* d_out, int out_size, void* d_ws, size_t ws_size,
                              hipStream_t stream) {
  (void)in_sizes; (void)n_in; (void)out_size; (void)ws_size;
  const float* noise = (const float*)d_in[0];
  const float* obj   = (const float*)d_in[1];
  const float* ln1w  = (const float*)d_in[2];
  const float* ln1b  = (const float*)d_in[3];
  const float* ln2w  = (const float*)d_in[4];
  const float* ln2b  = (const float*)d_in[5];
  const float* qw    = (const float*)d_in[6];
  const float* kvw   = (const float*)d_in[7];
  const float* W     = (const float*)d_in[8];
  const float* bng   = (const float*)d_in[9];
  const float* bnb   = (const float*)d_in[10];
  const float* projw = (const float*)d_in[11];
  const float* projb = (const float*)d_in[12];
  const float* fc1w  = (const float*)d_in[13];
  const float* fc1b  = (const float*)d_in[14];
  const float* fc2w  = (const float*)d_in[15];
  const float* fc2b  = (const float*)d_in[16];
  float* out = (float*)d_out;
  char* ws = (char*)d_ws;

  _Float16* qwh   = (_Float16*)(ws + OFF_QWH);
  _Float16* kvwh  = (_Float16*)(ws + OFF_KVWH);
  _Float16* Wth   = (_Float16*)(ws + OFF_WTH);
  _Float16* projh = (_Float16*)(ws + OFF_PROJ);
  _Float16* fc1h  = (_Float16*)(ws + OFF_FC1);
  _Float16* fc2h  = (_Float16*)(ws + OFF_FC2);
  float*    stats = (float*)(ws + OFF_STAT);
  _Float16* qh    = (_Float16*)(ws + OFF_QH);
  _Float16* kh    = (_Float16*)(ws + OFF_KH);
  _Float16* vt    = (_Float16*)(ws + OFF_VT);
  float*    attn2 = (float*)(ws + OFF_A2);
  float*    xresb = (float*)(ws + OFF_XR);

  prep_kernel<<<13312, 256, 0, stream>>>(qw, kvw, W, projw, fc1w, fc2w,
                                         qwh, kvwh, Wth, projh, fc1h, fc2h, stats);
  qproj_kernel<<<BATCH, 256, 0, stream>>>(noise, ln1w, ln1b, qwh, qh);
  kvproj_kernel<<<BATCH, 256, 0, stream>>>(obj, ln1w, ln1b, kvwh, kh, vt);
  attnW_kernel<<<BATCH, 256, 0, stream>>>(qh, kh, Wth, attn2, stats);
  bnattn_kernel<<<BATCH, 256, 0, stream>>>(attn2, stats, bng, bnb, vt, projh,
                                           projb, noise, xresb);
  mlp_kernel<<<BATCH, 256, 0, stream>>>(xresb, ln2w, ln2b, fc1h, fc1b,
                                        fc2h, fc2b, out);
}
